// Encoder_41343355191395
// MI455X (gfx1250) — compile-verified, matrix operand order checked by reading, not run on hardware
//
#include <hip/hip_runtime.h>
#include <hip/hip_bf16.h>
#include <math.h>

// CDNA5 / gfx1250 WMMA types
typedef __attribute__((ext_vector_type(16))) _Float16 v16h;
typedef __attribute__((ext_vector_type(8)))  float    v8f;

#define B_SZ 512
#define T_SZ 200
#define TT_N 13    // ceil(200/16) t-tiles
#define IN_D 88
#define IN_P 96    // 88 padded to multiple of 32 (K)
#define RN_D 60
#define RN_P 64    // 60 padded (both as K and as N)
#define Z_D  100
#define Z_P  128   // 100 padded as K
#define ZN_P 112   // 100 padded as N (7 tiles of 16)

// ---- WMMA fragment loaders (ISA 7.12.2 16-bit layouts) --------------------
// A (M=16 x K=32, f16): lanes 0-15 -> M=lane, K={kb..kb+7, kb+16..kb+23};
// lanes 16-31 -> M=lane-16, K offset by +8 within each 16-block.
__device__ __forceinline__ v16h ld_a(const _Float16* S, int ld, int kbase, int lane) {
  int m  = lane & 15;
  int kb = ((lane >> 4) << 3) + kbase;
  const _Float16* p = S + m * ld + kb;
  v16h a;
#pragma unroll
  for (int i = 0; i < 8; ++i) { a[i] = p[i]; a[8 + i] = p[16 + i]; }
  return a;
}
// B (K=32 x N=16, f16), weights staged as Wsh[n][k] (N x K = math-B transposed)
__device__ __forceinline__ v16h ld_b(const _Float16* W, int ld, int kbase, int nb, int lane) {
  int n  = nb + (lane & 15);
  int kb = ((lane >> 4) << 3) + kbase;
  const _Float16* p = W + n * ld + kb;
  v16h b;
#pragma unroll
  for (int i = 0; i < 8; ++i) { b[i] = p[i]; b[8 + i] = p[16 + i]; }
  return b;
}
__device__ __forceinline__ v8f wmma_f16(v16h a, v16h b, v8f c) {
  return __builtin_amdgcn_wmma_f32_16x16x32_f16(false, a, false, b, (short)0, c,
                                                false, false);
}

// ===========================================================================
// Kernel A: fully parallel input projection, grid = (t-tile, b): batch index
// is scalar, no per-lane div/mod. Full tiles take a branch-free store path
// with constant strides (folds to immediate VMEM offsets).
//   xp[t][b][n] = xrev[b][t][:] @ W_ih^T + b_ih + b_hh    (time-major output)
// ===========================================================================
__global__ __launch_bounds__(128) void xproj_kernel(
    const float* __restrict__ xrev, const float* __restrict__ W_ih,
    const float* __restrict__ b_ih, const float* __restrict__ b_hh,
    float* __restrict__ xp) {
  __shared__ _Float16 Wsh[RN_P * IN_P];
  __shared__ _Float16 xs[16 * IN_P];
  __shared__ float    bsh[RN_P];
  int tid = threadIdx.x;
  int t0  = blockIdx.x * 16;                    // 13 tiles; last one partial
  int b   = blockIdx.y;                         // scalar batch index
  for (int i = tid; i < RN_P * IN_P; i += 128) {
    int n = i / IN_P, k = i % IN_P;
    Wsh[i] = (n < RN_D && k < IN_D) ? (_Float16)W_ih[n * IN_D + k] : (_Float16)0.f;
  }
  for (int i = tid; i < RN_P; i += 128)
    bsh[i] = (i < RN_D) ? (b_ih[i] + b_hh[i]) : 0.f;
  for (int i = tid; i < 16 * IN_P; i += 128) {
    int m = i / IN_P, k = i % IN_P;
    bool ok = (t0 + m < T_SZ) && (k < IN_D);
    xs[i] = ok ? (_Float16)xrev[((size_t)b * T_SZ + t0 + m) * IN_D + k]
               : (_Float16)0.f;
  }
  __syncthreads();
  int wave = __builtin_amdgcn_readfirstlane(tid >> 5);
  int lane = tid & 31;
  int hi8  = (lane >> 4) << 3;
  int nb = wave * 16;
  int n  = nb + (lane & 15);
  float bv = bsh[n];
  v8f c = {bv, bv, bv, bv, bv, bv, bv, bv};
#pragma unroll
  for (int kk = 0; kk < IN_P; kk += 32)
    c = wmma_f16(ld_a(xs, IN_P, kk, lane), ld_b(Wsh, IN_P, kk, nb, lane), c);
  // C layout: VGPR v -> row M = v (+8 for hi lanes) -> t = t0 + v + hi8
  if (t0 + 16 <= T_SZ) {                        // full tile: branch-free path
    float* dst = xp + ((size_t)(t0 + hi8) * B_SZ + b) * RN_P + n;
#pragma unroll
    for (int v = 0; v < 8; ++v)
      dst[(size_t)v * (B_SZ * RN_P)] = c[v];    // constant 128KB strides
  } else {                                      // last tile: guarded path
#pragma unroll
    for (int v = 0; v < 8; ++v) {
      int t = t0 + v + hi8;
      if (t < T_SZ)
        xp[((size_t)t * B_SZ + b) * RN_P + n] = c[v];
    }
  }
}

// ===========================================================================
// Kernel B: sequential ReLU-RNN recurrence, software-pipelined.
//   h = relu(xp[t] + h @ W_hh^T); 32 blocks x 4 waves, 2 WMMAs/step.
//   Next step's xp tile prefetched (constant-stride, immediate offsets) under
//   the current WMMA chain; h exported as f16 via coalesced b128 stores.
// ===========================================================================
__global__ __launch_bounds__(128) void rnn_scan_kernel(
    const float* __restrict__ W_hh, const float* __restrict__ h_0,
    const float* __restrict__ xp, _Float16* __restrict__ rnn_out) {
  __shared__ _Float16 Wsh[RN_P * RN_P];
  __shared__ _Float16 hs[16 * RN_P];
  int tid = threadIdx.x;
  for (int i = tid; i < RN_P * RN_P; i += 128) {
    int n = i / RN_P, k = i % RN_P;
    Wsh[i] = (n < RN_D && k < RN_D) ? (_Float16)W_hh[n * RN_D + k] : (_Float16)0.f;
  }
  for (int i = tid; i < 16 * RN_P; i += 128) {
    int n = i % RN_P;
    hs[i] = (n < RN_D) ? (_Float16)h_0[n] : (_Float16)0.f;
  }
  __syncthreads();
  int wave = __builtin_amdgcn_readfirstlane(tid >> 5);
  int lane = tid & 31;
  int hi8  = (lane >> 4) << 3;
  int nb = wave * 16;
  int n  = nb + (lane & 15);
  int b0 = blockIdx.x * 16;
  // per-lane xp tile base for (t, rows b0+hi8..+7): advance by 128KB per step
  const float* xsrc = xp + ((size_t)(b0 + hi8)) * RN_P + n;
  // coalesced-export geometry: thread -> (row m, 8-half chunk ch) of hs
  int em = tid >> 3, ech = tid & 7;
  _Float16* erow = rnn_out + (size_t)(b0 + em) * T_SZ * RN_P + ech * 8;
  v8f cn;                                       // prefetched xp tile (t = 0)
#pragma unroll
  for (int v = 0; v < 8; ++v) cn[v] = xsrc[v * RN_P];
  for (int t = 0; t < T_SZ; ++t) {
    v8f c = cn;
    if (t + 1 < T_SZ) {                         // prefetch t+1 (indep of hs)
      const float* s1 = xsrc + (size_t)(t + 1) * (B_SZ * RN_P);
#pragma unroll
      for (int v = 0; v < 8; ++v) cn[v] = s1[v * RN_P];
    }
    c = wmma_f16(ld_a(hs, RN_P, 0, lane),  ld_b(Wsh, RN_P, 0, nb, lane),  c);
    c = wmma_f16(ld_a(hs, RN_P, 32, lane), ld_b(Wsh, RN_P, 32, nb, lane), c);
    __syncthreads();                            // all waves done reading hs
#pragma unroll
    for (int v = 0; v < 8; ++v)
      hs[(v + hi8) * RN_P + n] = (_Float16)fmaxf(c[v], 0.f);
    __syncthreads();                            // new h visible everywhere
    // coalesced export of the whole 16x64 f16 tile: 1 b128 store per thread
    float4 blk = *reinterpret_cast<const float4*>(hs + tid * 8);
    *reinterpret_cast<float4*>(erow + (size_t)t * RN_P) = blk;
  }
}

// ===========================================================================
// Kernel C: sequential combiner scan. 7 waves: waves 0-3 compute h_comb,
// all 7 waves own one loc/scale N-tile. eps + reversal-gather loads issued
// at the top of each step (independent of z) to overlap the WMMA chains.
// ===========================================================================
__global__ __launch_bounds__(224) void combiner_kernel(
    const int* __restrict__ slen, const float* __restrict__ eps,
    const float* __restrict__ z_q_0,
    const float* __restrict__ W_zh, const float* __restrict__ b_zh,
    const float* __restrict__ W_hl, const float* __restrict__ b_hl,
    const float* __restrict__ W_hs, const float* __restrict__ b_hs,
    const _Float16* __restrict__ rnn_out,
    float* __restrict__ out_z, float* __restrict__ out_loc,
    float* __restrict__ out_scale) {
  __shared__ _Float16 Wzh_sh[RN_P * Z_P];       // [64][128]  (N x K)
  __shared__ _Float16 Whl_sh[ZN_P * RN_P];      // [112][64]
  __shared__ _Float16 Whs_sh[ZN_P * RN_P];
  __shared__ _Float16 zs[16 * Z_P];
  __shared__ _Float16 hc[16 * RN_P];
  __shared__ float bzh_sh[RN_P];
  __shared__ float bhl_sh[ZN_P];
  __shared__ float bhs_sh[ZN_P];
  int tid = threadIdx.x;
  for (int i = tid; i < RN_P * Z_P; i += 224) {
    int n = i / Z_P, k = i % Z_P;
    Wzh_sh[i] = (n < RN_D && k < Z_D) ? (_Float16)W_zh[n * Z_D + k] : (_Float16)0.f;
  }
  for (int i = tid; i < ZN_P * RN_P; i += 224) {
    int n = i / RN_P, k = i % RN_P;
    bool ok = (n < Z_D && k < RN_D);
    Whl_sh[i] = ok ? (_Float16)W_hl[n * RN_D + k] : (_Float16)0.f;
    Whs_sh[i] = ok ? (_Float16)W_hs[n * RN_D + k] : (_Float16)0.f;
  }
  for (int i = tid; i < RN_P; i += 224) bzh_sh[i] = (i < RN_D) ? b_zh[i] : 0.f;
  for (int i = tid; i < ZN_P; i += 224) {
    bhl_sh[i] = (i < Z_D) ? b_hl[i] : 0.f;
    bhs_sh[i] = (i < Z_D) ? b_hs[i] : 0.f;
  }
  for (int i = tid; i < 16 * Z_P; i += 224) {
    int n = i % Z_P;
    zs[i] = (n < Z_D) ? (_Float16)z_q_0[n] : (_Float16)0.f;
  }
  __syncthreads();
  int wave = __builtin_amdgcn_readfirstlane(tid >> 5);  // 0..6, scalar
  int lane = tid & 31;
  int hi8  = (lane >> 4) << 3;
  int b0 = blockIdx.x * 16;
  int nT = wave * 16 + (lane & 15);             // this wave's column
  int sl_v[8];                                  // seq_lengths, hoisted
#pragma unroll
  for (int v = 0; v < 8; ++v) sl_v[v] = slen[b0 + v + hi8];
  // hoisted per-lane bases: eps (advance 200KB/step), outputs (advance 400B/step)
  const float* ebase = eps + ((size_t)(b0 + hi8)) * Z_D + nT;
  size_t obase = ((size_t)(b0 + hi8) * T_SZ) * Z_D + nT;
  for (int t = 0; t < T_SZ; ++t) {
    // ---- prefetch eps for this wave's phase-B tile (independent of z) ----
    float ev[8];
    {
      const float* ep = ebase + (size_t)t * (B_SZ * Z_D);
#pragma unroll
      for (int v = 0; v < 8; ++v)
        ev[v] = (nT < Z_D) ? ep[v * Z_D] : 0.f; // constant 400B strides
    }
    // ---- phase A (waves 0-3): h_comb = 0.5*(tanh(z@Wzh^T+b) + rnn_rev_t) --
    if (wave < 4) {
      float ht[8];                              // reversal gather, issued first
#pragma unroll
      for (int v = 0; v < 8; ++v) {
        int rid = sl_v[v] - 1 - t;              // data-dependent index
        ht[v] = (rid >= 0)
              ? (float)rnn_out[((size_t)(b0 + v + hi8) * T_SZ + rid) * RN_P + nT]
              : 0.f;
      }
      float bv = bzh_sh[nT];
      v8f c = {bv, bv, bv, bv, bv, bv, bv, bv};
      int nb = wave * 16;
#pragma unroll
      for (int kk = 0; kk < Z_P; kk += 32)
        c = wmma_f16(ld_a(zs, Z_P, kk, lane), ld_b(Wzh_sh, Z_P, kk, nb, lane), c);
#pragma unroll
      for (int v = 0; v < 8; ++v)
        hc[(v + hi8) * RN_P + nT] = (_Float16)(0.5f * (tanhf(c[v]) + ht[v]));
    }
    __syncthreads();                            // hc ready; zs reads done
    // ---- phase B (waves 0-6): one loc/scale N-tile per wave ----
    {
      int nb = wave * 16;
      float bl = bhl_sh[nT], bs = bhs_sh[nT];
      v8f cl = {bl, bl, bl, bl, bl, bl, bl, bl};
      v8f cs = {bs, bs, bs, bs, bs, bs, bs, bs};
      v16h a0 = ld_a(hc, RN_P, 0, lane);
      v16h a1 = ld_a(hc, RN_P, 32, lane);
      cl = wmma_f16(a0, ld_b(Whl_sh, RN_P, 0, nb, lane),  cl);
      cl = wmma_f16(a1, ld_b(Whl_sh, RN_P, 32, nb, lane), cl);
      cs = wmma_f16(a0, ld_b(Whs_sh, RN_P, 0, nb, lane),  cs);
      cs = wmma_f16(a1, ld_b(Whs_sh, RN_P, 32, nb, lane), cs);
      size_t ot = obase + (size_t)t * Z_D;      // row base for this step
#pragma unroll
      for (int v = 0; v < 8; ++v) {
        if (nT < Z_D) {
          float loc = cl[v];
          float sc  = cs[v];
          sc = (sc > 20.f) ? sc : log1pf(__expf(sc));     // softplus
          float z = loc + sc * ev[v];
          size_t o = ot + (size_t)v * (T_SZ * Z_D);       // constant strides
          out_z[o] = z; out_loc[o] = loc; out_scale[o] = sc;
          zs[(v + hi8) * Z_P + nT] = (_Float16)z;         // recycle z
        }
      }
    }
    __syncthreads();                            // zs updated for next step
  }
}

extern "C" void kernel_launch(void* const* d_in, const int* in_sizes, int n_in,
                              void* d_out, int out_size, void* d_ws, size_t ws_size,
                              hipStream_t stream) {
  const float* xrev  = (const float*)d_in[1];   // mini_batch_reversed [B,T,88]
  const int*   slen  = (const int*)  d_in[3];   // seq lengths [B]
  const float* eps   = (const float*)d_in[4];   // [T,B,100]
  const float* W_ih  = (const float*)d_in[5];
  const float* W_hh  = (const float*)d_in[6];
  const float* b_ih  = (const float*)d_in[7];
  const float* b_hh  = (const float*)d_in[8];
  const float* h_0   = (const float*)d_in[9];
  const float* z_q_0 = (const float*)d_in[10];
  const float* W_zh  = (const float*)d_in[11];
  const float* b_zh  = (const float*)d_in[12];
  const float* W_hl  = (const float*)d_in[13];
  const float* b_hl  = (const float*)d_in[14];
  const float* W_hs  = (const float*)d_in[15];
  const float* b_hs  = (const float*)d_in[16];

  // workspace: xp f32 [T][B][64] (26.2MB) then rnn_out f16 [B][T][64] (13.1MB)
  float*     xp      = (float*)d_ws;
  _Float16*  rnn_out = (_Float16*)(xp + (size_t)T_SZ * B_SZ * RN_P);

  float* out_z     = (float*)d_out;
  float* out_loc   = out_z   + (size_t)B_SZ * T_SZ * Z_D;
  float* out_scale = out_loc + (size_t)B_SZ * T_SZ * Z_D;

  hipLaunchKernelGGL(xproj_kernel, dim3(TT_N, B_SZ), dim3(128), 0, stream,
                     xrev, W_ih, b_ih, b_hh, xp);
  hipLaunchKernelGGL(rnn_scan_kernel, dim3(B_SZ / 16), dim3(128), 0, stream,
                     W_hh, h_0, xp, rnn_out);
  hipLaunchKernelGGL(combiner_kernel, dim3(B_SZ / 16), dim3(224), 0, stream,
                     slen, eps, z_q_0, W_zh, b_zh, W_hl, b_hl, W_hs, b_hs,
                     rnn_out, out_z, out_loc, out_scale);
}
